// Relation_Attention_Diff_52682068853337
// MI455X (gfx1250) — compile-verified, matrix operand order checked by reading, not run on hardware
//
#include <hip/hip_runtime.h>
#include <hip/hip_bf16.h>

// Problem constants (from reference)
#define B_   16
#define C_   128
#define C8_  16
#define H_   96
#define W_   96
#define HW_  (H_ * W_)                       // 9216
#define OUTSZ_ ((size_t)B_ * C_ * HW_)       // 18,874,368 per output tensor

typedef __attribute__((ext_vector_type(16))) __bf16 bf16x16;
typedef __attribute__((ext_vector_type(8)))  float  v8f;

// float -> bf16 bits using the native converter (RNE on gfx1250)
__device__ __forceinline__ unsigned short bf_bits(float f) {
    return __builtin_bit_cast(unsigned short, (__bf16)f);
}

// Per-lane bf16 fragment = two contiguous 8-element K runs:
//   us[0..7]  <- mem[k0 + khalf*8 + 0..7]
//   us[8..15] <- mem[k0 + khalf*8 + 16..23]
union Frag {
    bf16x16        v;
    uint4          q2[2];
    unsigned short us[16];
};

__device__ __forceinline__ bf16x16 frag_bf16_contig(const unsigned short* p8) {
    Frag f;
    f.q2[0] = *(const uint4*)(p8);
    f.q2[1] = *(const uint4*)(p8 + 16);
    return f.v;
}
// K=16 case: only the first half is real, upper 16 K's are zero padding.
__device__ __forceinline__ bf16x16 frag_bf16_k16(const unsigned short* p8) {
    Frag f;
    f.q2[0] = *(const uint4*)(p8);
    uint4 z; z.x = 0u; z.y = 0u; z.z = 0u; z.w = 0u;
    f.q2[1] = z;
    return f.v;
}

__device__ __forceinline__ v8f wmma_bf16(bf16x16 a, bf16x16 b, v8f c) {
    return __builtin_amdgcn_wmma_f32_16x16x32_bf16(false, a, false, b, (short)0, c,
                                                   false, false);
}

// ---------------------------------------------------------------------------
// Kernel P: one-shot weight conversion fp32 -> bf16 (36,864 elements total).
// Packed buffer: [Wq 16x128 | Wk 16x128 | Wv 128x128 | Wqv 128x128]
// ---------------------------------------------------------------------------
#define WQ_OFF_  0
#define WK_OFF_  (C8_ * C_)
#define WV_OFF_  (2 * C8_ * C_)
#define WQV_OFF_ (2 * C8_ * C_ + C_ * C_)
#define WTOT_    (2 * C8_ * C_ + 2 * C_ * C_)        // 36,864

__global__ void wcvt_kernel(const float* __restrict__ Wq, const float* __restrict__ Wk,
                            const float* __restrict__ Wv, const float* __restrict__ Wqv,
                            unsigned short* __restrict__ wb)
{
    int i = blockIdx.x * 256 + threadIdx.x;
    float f;
    if      (i < WK_OFF_)  f = Wq [i];
    else if (i < WV_OFF_)  f = Wk [i - WK_OFF_];
    else if (i < WQV_OFF_) f = Wv [i - WV_OFF_];
    else                   f = Wqv[i - WQV_OFF_];
    wb[i] = bf_bits(f);
}

// ---------------------------------------------------------------------------
// Kernel 0: tiled LDS transpose + fp32->bf16: x (C,HW) -> xT (HW,C) per batch.
// grid: (HW/32, C/32, 2*B), block (32,8).  which=0: xe->xTe, 1: xq->xTq.
// ---------------------------------------------------------------------------
__global__ void transpose_kernel(const float* __restrict__ xe, const float* __restrict__ xq,
                                 unsigned short* __restrict__ xTe, unsigned short* __restrict__ xTq)
{
    __shared__ unsigned short tile[32][33];
    const int which = blockIdx.z & 1;
    const int b     = blockIdx.z >> 1;
    const float* src    = (which ? xq  : xe)  + (size_t)b * C_ * HW_;
    unsigned short* dst = (which ? xTq : xTe) + (size_t)b * HW_ * C_;
    const int p0 = blockIdx.x * 32;           // position tile
    const int c0 = blockIdx.y * 32;           // channel tile

#pragma unroll
    for (int i = 0; i < 4; ++i) {
        int c = threadIdx.y + i * 8;
        tile[c][threadIdx.x] = bf_bits(src[(size_t)(c0 + c) * HW_ + p0 + threadIdx.x]);
    }
    __syncthreads();
#pragma unroll
    for (int i = 0; i < 4; ++i) {
        int p = threadIdx.y + i * 8;
        dst[(size_t)(p0 + p) * C_ + c0 + threadIdx.x] = tile[threadIdx.x][p];
    }
}

// ---------------------------------------------------------------------------
// Kernel 1: all four projections. GEMM orientation: M=out-channel, N=position,
// K=C. Both operands K-contiguous bf16 => all fragments are 2 x b128 loads.
//  q,k  -> position-major (HW,16): one packed b128 store per lane
//  v,qv -> both (C,H,W) and (C,W,H) layouts (scalar b16 stores)
// grid: (72, 18, B), block 256 (8 waves).
// ---------------------------------------------------------------------------
__global__ void proj_kernel(const unsigned short* __restrict__ xTq,
                            const unsigned short* __restrict__ xTe,
                            const unsigned short* __restrict__ wb,
                            const float* __restrict__ bq,  const float* __restrict__ bk,
                            const float* __restrict__ bv,  const float* __restrict__ bqv,
                            unsigned short* __restrict__ qT, unsigned short* __restrict__ kT,
                            unsigned short* __restrict__ v,  unsigned short* __restrict__ vH,
                            unsigned short* __restrict__ qv, unsigned short* __restrict__ qvH)
{
    const int lane = threadIdx.x & 31;
    const int wave = threadIdx.x >> 5;
    const int n0   = (blockIdx.x * 8 + wave) * 16;    // position tile start
    const int y    = blockIdx.y;
    const int b    = blockIdx.z;

    const unsigned short* xT; const unsigned short* Wm; const float* bias; int mtile; int kind;
    if (y == 0)      { xT = xTq; Wm = wb + WQ_OFF_;  bias = bq;  mtile = 0;      kind = 0; }
    else if (y == 1) { xT = xTe; Wm = wb + WK_OFF_;  bias = bk;  mtile = 0;      kind = 1; }
    else if (y < 10) { xT = xTe; Wm = wb + WV_OFF_;  bias = bv;  mtile = y - 2;  kind = 2; }
    else             { xT = xTq; Wm = wb + WQV_OFF_; bias = bqv; mtile = y - 10; kind = 3; }
    xT += (size_t)b * HW_ * C_;

    const int nl = lane & 15, khalf = lane >> 4;
    const int mA   = mtile * 16 + nl;                 // out-channel row (A)
    const int posn = n0 + nl;                         // position col (B)

    v8f acc = {};
    for (int k0 = 0; k0 < C_; k0 += 32) {
        bf16x16 a  = frag_bf16_contig(Wm + (size_t)mA * C_ + k0 + khalf * 8);
        bf16x16 bm = frag_bf16_contig(xT + (size_t)posn * C_ + k0 + khalf * 8);
        acc = wmma_bf16(a, bm, acc);
    }

    if (kind < 2) {
        // D: lane=position, r -> channel khalf*8+r (consecutive) => packed store
        unsigned short* dst = (kind == 0 ? qT : kT) + (size_t)b * HW_ * C8_;
        Frag f;
#pragma unroll
        for (int r = 0; r < 8; ++r) f.us[r] = bf_bits(acc[r] + bias[khalf * 8 + r]);
        *(uint4*)(dst + (size_t)posn * C8_ + khalf * 8) = f.q2[0];
    } else {
        unsigned short* d1 = (kind == 2 ? v  : qv ) + (size_t)b * C_ * HW_;
        unsigned short* d2 = (kind == 2 ? vH : qvH) + (size_t)b * C_ * HW_;
        const int h = posn / W_, w = posn % W_;
#pragma unroll
        for (int r = 0; r < 8; ++r) {
            int oc = mtile * 16 + khalf * 8 + r;
            unsigned short bv16 = bf_bits(acc[r] + bias[oc]);
            d1[(size_t)oc * HW_ + posn]       = bv16;     // (C,H,W)
            d2[(size_t)oc * HW_ + w * H_ + h] = bv16;     // (C,W,H)
        }
    }
}

// ---------------------------------------------------------------------------
// Kernel 2: scores. mode 0: eH (fixed w), mode 1: eW (fixed h). K=16 padded.
// qT/kT are position-major => A/B fragments are single b128 loads.
// grid: (9, 96, 2*B), block 128. e layout: (B,H,W,192) fp32.
// ---------------------------------------------------------------------------
__global__ void score_kernel(const unsigned short* __restrict__ qT_all,
                             const unsigned short* __restrict__ kT_all,
                             float* __restrict__ e)
{
    const int lane = threadIdx.x & 31;
    const int wave = threadIdx.x >> 5;
    const int t     = blockIdx.x * 4 + wave;          // 0..35
    const int mtile = t / 6, ntile = t % 6;
    const int f     = blockIdx.y;
    const int mode  = blockIdx.z & 1;
    const int b     = blockIdx.z >> 1;

    const unsigned short* qT = qT_all + (size_t)b * HW_ * C8_;
    const unsigned short* kT = kT_all + (size_t)b * HW_ * C8_;

    const int stride = (mode == 0) ? W_ : 1;
    const int foff   = (mode == 0) ? f  : f * W_;
    const int nl = lane & 15, khalf = lane >> 4;
    const int mA = mtile * 16 + nl;
    const int nB = ntile * 16 + nl;

    bf16x16 a  = frag_bf16_k16(qT + (size_t)(foff + mA * stride) * C8_ + khalf * 8);
    bf16x16 bm = frag_bf16_k16(kT + (size_t)(foff + nB * stride) * C8_ + khalf * 8);
    v8f acc = {};
    acc = wmma_bf16(a, bm, acc);

    const int coloff = mode ? 96 : 0;
#pragma unroll
    for (int r = 0; r < 8; ++r) {
        int mm = mtile * 16 + r + khalf * 8;
        size_t row = (mode == 0) ? ((size_t)(b * H_ + mm) * W_ + f)
                                 : ((size_t)(b * H_ + f) * W_ + mm);
        e[row * 192 + coloff + ntile * 16 + nl] = acc[r];
    }
}

// ---------------------------------------------------------------------------
// Kernel 3: joint softmax over 192 values/row; diag (g==h) masked; bf16 out.
// One wave per row; block 256 = 8 rows. Rows = B*H*W = 147456.
// ---------------------------------------------------------------------------
__global__ void softmax_kernel(const float* __restrict__ e, unsigned short* __restrict__ att)
{
    const int lane = threadIdx.x & 31;
    const int wave = threadIdx.x >> 5;
    const size_t row = (size_t)blockIdx.x * 8 + wave;
    const int h = (int)((row / W_) % H_);
    const float* er = e + row * 192;

    float vals[6];
    float mx = -1e30f;
#pragma unroll
    for (int j = 0; j < 6; ++j) {
        int idx = lane + 32 * j;
        float x = er[idx];
        if (idx == h) x = -1e30f;   // diag mask (eH half: idx<96, h<96)
        vals[j] = x;
        mx = fmaxf(mx, x);
    }
#pragma unroll
    for (int off = 16; off; off >>= 1) mx = fmaxf(mx, __shfl_xor(mx, off, 32));

    float s = 0.f;
#pragma unroll
    for (int j = 0; j < 6; ++j) { vals[j] = __expf(vals[j] - mx); s += vals[j]; }
#pragma unroll
    for (int off = 16; off; off >>= 1) s += __shfl_xor(s, off, 32);
    const float inv = 1.0f / s;

#pragma unroll
    for (int j = 0; j < 6; ++j)
        att[row * 192 + lane + 32 * j] = bf_bits(vals[j] * inv);
}

// ---------------------------------------------------------------------------
// Kernel 4a: outH/qoutH per (b,w): M=c(128), N=h(96), K=g(96).
// vH (C,W,H) + att rows are K-contiguous => all fragments are b128 loads.
// Fused epilogue (LR branch == +2 exactly): out = gamma*(acc+2) + x.
// grid: (12, 96, B), block 128.
// ---------------------------------------------------------------------------
__global__ void outH_kernel(const unsigned short* __restrict__ vH_all,
                            const unsigned short* __restrict__ qvH_all,
                            const unsigned short* __restrict__ att,
                            const float* __restrict__ xe, const float* __restrict__ xq,
                            const float* __restrict__ g1p, const float* __restrict__ g2p,
                            float* __restrict__ out)
{
    const int lane = threadIdx.x & 31;
    const int wave = threadIdx.x >> 5;
    const int t     = blockIdx.x * 4 + wave;          // 0..47
    const int mtile = t / 6, ntile = t % 6;
    const int w = blockIdx.y, b = blockIdx.z;
    const unsigned short* vH  = vH_all  + (size_t)b * C_ * HW_;
    const unsigned short* qvH = qvH_all + (size_t)b * C_ * HW_;

    const int nl = lane & 15, khalf = lane >> 4;
    const int mA = mtile * 16 + nl;                   // c
    const int nB = ntile * 16 + nl;                   // h

    v8f acc1 = {}, acc2 = {};
    for (int k0 = 0; k0 < H_; k0 += 32) {
        bf16x16 a1 = frag_bf16_contig(vH  + (size_t)mA * HW_ + w * H_ + k0 + khalf * 8);
        bf16x16 a2 = frag_bf16_contig(qvH + (size_t)mA * HW_ + w * H_ + k0 + khalf * 8);
        bf16x16 bm = frag_bf16_contig(att + ((size_t)(b * H_ + nB) * W_ + w) * 192 + k0 + khalf * 8);
        acc1 = wmma_bf16(a1, bm, acc1);
        acc2 = wmma_bf16(a2, bm, acc2);
    }
    const float g1 = g1p[0], g2 = g2p[0];
#pragma unroll
    for (int r = 0; r < 8; ++r) {
        int c = mtile * 16 + r + khalf * 8;
        int h = ntile * 16 + nl;
        size_t idx = (((size_t)b * C_ + c) * H_ + h) * W_ + w;
        out[idx]          = g1 * (acc1[r] + 2.0f) + xe[idx];
        out[OUTSZ_ + idx] = g2 * (acc2[r] + 2.0f) + xq[idx];
    }
}

// ---------------------------------------------------------------------------
// Kernel 4b: outW/qoutW per (b,h): M=c, N=w, K=v(96). out += gamma*acc.
// v (C,H,W) + att rows K-contiguous => b128 fragment loads; coalesced RMW.
// grid: (12, 96, B), block 128.
// ---------------------------------------------------------------------------
__global__ void outW_kernel(const unsigned short* __restrict__ v_all,
                            const unsigned short* __restrict__ qv_all,
                            const unsigned short* __restrict__ att,
                            const float* __restrict__ g1p, const float* __restrict__ g2p,
                            float* __restrict__ out)
{
    const int lane = threadIdx.x & 31;
    const int wave = threadIdx.x >> 5;
    const int t     = blockIdx.x * 4 + wave;          // 0..47
    const int mtile = t / 6, ntile = t % 6;
    const int h = blockIdx.y, b = blockIdx.z;
    const unsigned short* v  = v_all  + (size_t)b * C_ * HW_;
    const unsigned short* qv = qv_all + (size_t)b * C_ * HW_;

    const int nl = lane & 15, khalf = lane >> 4;
    const int mA = mtile * 16 + nl;                   // c
    const int nB = ntile * 16 + nl;                   // w

    v8f acc1 = {}, acc2 = {};
    for (int k0 = 0; k0 < W_; k0 += 32) {
        bf16x16 a1 = frag_bf16_contig(v   + (size_t)mA * HW_ + h * W_ + k0 + khalf * 8);
        bf16x16 a2 = frag_bf16_contig(qv  + (size_t)mA * HW_ + h * W_ + k0 + khalf * 8);
        bf16x16 bm = frag_bf16_contig(att + ((size_t)(b * H_ + h) * W_ + nB) * 192 + 96 + k0 + khalf * 8);
        acc1 = wmma_bf16(a1, bm, acc1);
        acc2 = wmma_bf16(a2, bm, acc2);
    }
    const float g1 = g1p[0], g2 = g2p[0];
#pragma unroll
    for (int r = 0; r < 8; ++r) {
        int c = mtile * 16 + r + khalf * 8;
        int wcol = ntile * 16 + nl;
        size_t idx = (((size_t)b * C_ + c) * H_ + h) * W_ + wcol;
        out[idx]          += g1 * acc1[r];
        out[OUTSZ_ + idx] += g2 * acc2[r];
    }
}

// ---------------------------------------------------------------------------
extern "C" void kernel_launch(void* const* d_in, const int* in_sizes, int n_in,
                              void* d_out, int out_size, void* d_ws, size_t ws_size,
                              hipStream_t stream)
{
    const float* xe  = (const float*)d_in[0];
    const float* xq  = (const float*)d_in[1];
    const float* Wq  = (const float*)d_in[2];
    const float* bq  = (const float*)d_in[3];
    const float* Wk  = (const float*)d_in[4];
    const float* bk  = (const float*)d_in[5];
    const float* Wv  = (const float*)d_in[6];
    const float* bv  = (const float*)d_in[7];
    const float* Wqv = (const float*)d_in[8];
    const float* bqv = (const float*)d_in[9];
    const float* g1  = (const float*)d_in[10];
    const float* g2  = (const float*)d_in[11];
    float* out = (float*)d_out;

    // Workspace carve-up (~330 MB). xT region is reused for fp32 scores (e):
    // xT is only read by proj (kernel 1), e is only written from kernel 2 on.
    const size_t XT = (size_t)B_ * HW_ * C_  * 2;   // 37,748,736 B (bf16)
    const size_t QT = (size_t)B_ * HW_ * C8_ * 2;   //  4,718,592 B
    const size_t VB = XT;
    const size_t ES = (size_t)B_ * HW_ * 192 * 4;   // 113,246,208 B (fp32)
    const size_t AT = (size_t)B_ * HW_ * 192 * 2;   //  56,623,104 B (bf16)

    char* p = (char*)d_ws;
    unsigned short* xTe = (unsigned short*)p;
    unsigned short* xTq = (unsigned short*)(p + XT);
    float*          e   = (float*)p;                 // aliases xTe/xTq region
    p += (ES > 2 * XT ? ES : 2 * XT);
    unsigned short* qT  = (unsigned short*)p; p += QT;
    unsigned short* kT  = (unsigned short*)p; p += QT;
    unsigned short* v   = (unsigned short*)p; p += VB;
    unsigned short* vH  = (unsigned short*)p; p += VB;
    unsigned short* qv  = (unsigned short*)p; p += VB;
    unsigned short* qvH = (unsigned short*)p; p += VB;
    unsigned short* att = (unsigned short*)p; p += AT;
    unsigned short* wb  = (unsigned short*)p; p += (size_t)WTOT_ * 2;
    (void)ws_size; (void)in_sizes; (void)n_in; (void)out_size;

    // P) weights fp32 -> bf16 (once; 36,864 elements)
    wcvt_kernel<<<dim3(WTOT_ / 256), 256, 0, stream>>>(Wq, Wk, Wv, Wqv, wb);

    // 0) transpose + convert inputs: x (C,HW) fp32 -> xT (HW,C) bf16
    transpose_kernel<<<dim3(HW_ / 32, C_ / 32, 2 * B_), dim3(32, 8), 0, stream>>>(
        xe, xq, xTe, xTq);

    // 1) projections (qT,kT position-major; v,qv in two layouts)
    proj_kernel<<<dim3(HW_ / (8 * 16), 18, B_), 256, 0, stream>>>(
        xTq, xTe, wb, bq, bk, bv, bqv, qT, kT, v, vH, qv, qvH);

    // 2) scores eH + eW -> fp32 (B,H,W,192)  (overwrites xT region)
    score_kernel<<<dim3(9, 96, 2 * B_), 128, 0, stream>>>(qT, kT, e);

    // 3) joint masked softmax -> bf16 att
    softmax_kernel<<<dim3((B_ * HW_) / 8), 256, 0, stream>>>(e, att);

    // 4a) H-direction output GEMMs + fused epilogue (writes d_out)
    outH_kernel<<<dim3(12, W_, B_), 128, 0, stream>>>(vH, qvH, att, xe, xq, g1, g2, out);

    // 4b) W-direction output GEMMs (accumulates into d_out)
    outW_kernel<<<dim3(12, H_, B_), 128, 0, stream>>>(v, qv, att, g1, g2, out);
}
